// TCRGTransform_44470091383135
// MI455X (gfx1250) — matrix operand order repaired and checked by reading, not compiled, not run on hardware
//
#include <hip/hip_runtime.h>
#include <hip/hip_bf16.h>

typedef _Float16 f16;
typedef __attribute__((ext_vector_type(16))) _Float16 v16h;
typedef __attribute__((ext_vector_type(8)))  _Float16 v8h;
typedef __attribute__((ext_vector_type(8)))  float    v8f;

static constexpr int N_NODES   = 20000;
static constexpr int N_EDGES   = 150000;
static constexpr int NODE_DIM  = 32;
static constexpr int EDGE_DIM  = 8;
static constexpr int HIDDEN    = 64;
static constexpr int OUT_DIM   = 64;
static constexpr int N_GRAPHS  = 64;
static constexpr int MLP_HID   = 128;

static constexpr int KTILES    = 129;   // 128 h-steps + 1 bias(b_fc2) step; K = 129*32 = 4128
static constexpr int HSTRIDE   = 132;   // padded f16 row stride for per-edge h
static constexpr int NT        = 4;     // 64 output cols / 16
static constexpr int EDGE_TILES = (N_EDGES + 15) / 16;  // 9375
static constexpr int GEMM_WAVES = 4;
static constexpr int GEMM_T     = 2;    // 16-edge M-tiles per wave

// ---------------------------------------------------------------- utilities
__global__ void k_zero(float* p, int n) {
  int i = blockIdx.x * blockDim.x + threadIdx.x;
  int stride = gridDim.x * blockDim.x;
  for (; i < n; i += stride) p[i] = 0.0f;
}

__global__ void k_x_to_f16(const float* __restrict__ x, f16* __restrict__ xh, int n) {
  int i = blockIdx.x * blockDim.x + threadIdx.x;
  if (i < n) xh[i] = (f16)x[i];
}

// Repack W_fc2 (+ b_fc2 as the last 32 K-rows) into WMMA-B fragment layout.
// Fragment f = kt*4 + nt; lane holds 16 consecutive-K f16 (32 contiguous bytes):
//   col o = nt*16 + (lane&15), K = kt*32 + (lane&16) + j, j = 0..15
__global__ void k_build_bfrag(const float* __restrict__ W_fc2,
                              const float* __restrict__ b_fc2,
                              f16* __restrict__ Bfrag) {
  int t = blockIdx.x * blockDim.x + threadIdx.x;   // one thread = one (frag, lane)
  if (t >= KTILES * NT * 32) return;
  int lane = t & 31;
  int frag = t >> 5;
  int nt = frag & 3;
  int kt = frag >> 2;
  int o  = nt * 16 + (lane & 15);
  int kbase = kt * 32 + (lane & 16);
  v16h v;
  #pragma unroll
  for (int j = 0; j < 16; ++j) {
    int kg = kbase + j;                  // global K in [0, 4128)
    float val;
    if (kg < 4096) {                     // kg = m*32 + i
      int m = kg >> 5, i = kg & 31;
      val = W_fc2[m * 2048 + i * 64 + o];
    } else {                             // bias rows, paired with h == 1
      val = b_fc2[(kg - 4096) * 64 + o];
    }
    v[j] = (f16)val;
  }
  *(((v16h*)Bfrag) + t) = v;
}

// Edge MLP layer 1 (relu(edge_attr @ W_fc1 + b_fc1)) -> f16, plus the constant-1
// bias lane, plus per-destination degree counting. One block per edge.
__global__ void k_edge_fc1(const float* __restrict__ edge_attr,
                           const int*   __restrict__ edge_index,
                           const float* __restrict__ W_fc1,
                           const float* __restrict__ b_fc1,
                           f16* __restrict__ h16, float* __restrict__ cnt) {
  __shared__ float ea[EDGE_DIM];
  int e = blockIdx.x;
  int t = threadIdx.x;
  if (t < EDGE_DIM) ea[t] = edge_attr[e * EDGE_DIM + t];
  __syncthreads();
  float acc = b_fc1[t];
  #pragma unroll
  for (int i = 0; i < EDGE_DIM; ++i) acc = fmaf(ea[i], W_fc1[i * MLP_HID + t], acc);
  acc = fmaxf(acc, 0.0f);
  size_t row = (size_t)e * HSTRIDE;
  h16[row + t] = (f16)acc;
  if (t == 0) {
    h16[row + MLP_HID] = (f16)1.0f;            // bias K-block multiplier
    atomicAdd(&cnt[edge_index[N_EDGES + e]], 1.0f);
  }
}

// Fused GEMM: msg[E,64] = P[E,4128] @ B[4128,64], P built in registers.
// A fragment per K-step = (constant x-fragment) * h[e, kt]  (v_pk_mul_f16 x8).
__global__ void __launch_bounds__(GEMM_WAVES * 32)
k_edge_gemm(const f16* __restrict__ xh, const f16* __restrict__ h16,
            const f16* __restrict__ Bfrag, const int* __restrict__ edge_index,
            float* __restrict__ aggr) {
  const int lane = threadIdx.x & 31;
  const int wave = threadIdx.x >> 5;
  const int r  = lane & 15;
  const int hi = lane >> 4;

  v8f  acc[GEMM_T][NT];
  v16h axf[GEMM_T];
  const f16* hrow[GEMM_T];
  bool valid[GEMM_T];
  int  e0[GEMM_T];
  v8f zero = {};

  #pragma unroll
  for (int mt = 0; mt < GEMM_T; ++mt) {
    int tile = (blockIdx.x * GEMM_WAVES + wave) * GEMM_T + mt;
    valid[mt] = (tile < EDGE_TILES);
    if (tile >= EDGE_TILES) tile = EDGE_TILES - 1;   // clamp; scatter is guarded
    e0[mt] = tile * 16;
    int e = e0[mt] + r;
    int s = edge_index[e];                           // source node
    const f16* xb = xh + (size_t)s * NODE_DIM;
    // 16-bit A 16x32 layout (contiguous, matches Bfrag): lane<16 holds K {0..15}, lane>=16 {16..31}
    axf[mt] = *(const v16h*)(xb + hi * 16);
    hrow[mt] = h16 + (size_t)e * HSTRIDE;
    #pragma unroll
    for (int nt = 0; nt < NT; ++nt) acc[mt][nt] = zero;
  }

  const v16h* Bf = (const v16h*)Bfrag;
  for (int kt = 0; kt < KTILES; ++kt) {
    v16h bf[NT];
    #pragma unroll
    for (int nt = 0; nt < NT; ++nt) bf[nt] = Bf[(kt * NT + nt) * 32 + lane];
    #pragma unroll
    for (int mt = 0; mt < GEMM_T; ++mt) {
      f16 hv = hrow[mt][kt];            // per-row scale of the x fragment
      v16h a = axf[mt] * hv;
      #pragma unroll
      for (int nt = 0; nt < NT; ++nt) {
        acc[mt][nt] = __builtin_amdgcn_wmma_f32_16x16x32_f16(
            false, a, false, bf[nt], (short)0, acc[mt][nt], false, false);
      }
    }
  }

  // Scatter-add msg rows into aggr[dst]. D layout: VGPR v -> row v + 8*hi, col lane&15.
  #pragma unroll
  for (int mt = 0; mt < GEMM_T; ++mt) {
    if (!valid[mt]) continue;
    #pragma unroll
    for (int v = 0; v < 8; ++v) {
      int e = e0[mt] + v + hi * 8;
      int dst = edge_index[N_EDGES + e];
      float* arow = aggr + (size_t)dst * HIDDEN;
      #pragma unroll
      for (int nt = 0; nt < NT; ++nt)
        atomicAdd(&arow[nt * 16 + (lane & 15)], acc[mt][nt][v]);
    }
  }
}

// Per-node: mean-aggregate, + x@W_root + b_conv, relu, @W_pre + b_pre, relu,
// then pool into per-graph sums. One block (64 threads) per node.
__global__ void k_node(const float* __restrict__ x, const int* __restrict__ batch,
                       const float* __restrict__ aggr, const float* __restrict__ cnt,
                       const float* __restrict__ W_root, const float* __restrict__ b_conv,
                       const float* __restrict__ W_pre,  const float* __restrict__ b_pre,
                       float* __restrict__ gsum, float* __restrict__ gcnt) {
  __shared__ float xs[NODE_DIM];
  __shared__ float v1s[HIDDEN];
  int n = blockIdx.x;
  int t = threadIdx.x;
  if (t < NODE_DIM) xs[t] = x[n * NODE_DIM + t];
  __syncthreads();
  float a = aggr[(size_t)n * HIDDEN + t] / fmaxf(cnt[n], 1.0f);
  float rt = b_conv[t];
  #pragma unroll
  for (int i = 0; i < NODE_DIM; ++i) rt = fmaf(xs[i], W_root[i * HIDDEN + t], rt);
  float v1 = fmaxf(a + rt, 0.0f);
  v1s[t] = v1;
  __syncthreads();
  float v2 = b_pre[t];
  #pragma unroll
  for (int j = 0; j < HIDDEN; ++j) v2 = fmaf(v1s[j], W_pre[j * HIDDEN + t], v2);
  v2 = fmaxf(v2, 0.0f);
  int b = batch[n];
  atomicAdd(&gsum[b * HIDDEN + t], v2);
  if (t == 0) atomicAdd(&gcnt[b], 1.0f);
}

// Per-graph: mean pool then relu(g @ W_out + b_out).
__global__ void k_graph_out(const float* __restrict__ gsum, const float* __restrict__ gcnt,
                            const float* __restrict__ W_out, const float* __restrict__ b_out,
                            float* __restrict__ out) {
  __shared__ float gs[HIDDEN];
  int g = blockIdx.x, t = threadIdx.x;
  gs[t] = gsum[g * HIDDEN + t] / fmaxf(gcnt[g], 1.0f);
  __syncthreads();
  float v = b_out[t];
  #pragma unroll
  for (int j = 0; j < HIDDEN; ++j) v = fmaf(gs[j], W_out[j * OUT_DIM + t], v);
  out[g * OUT_DIM + t] = fmaxf(v, 0.0f);
}

// ---------------------------------------------------------------- launch
extern "C" void kernel_launch(void* const* d_in, const int* in_sizes, int n_in,
                              void* d_out, int out_size, void* d_ws, size_t ws_size,
                              hipStream_t stream) {
  const float* x          = (const float*)d_in[0];
  const int*   edge_index = (const int*)  d_in[1];
  const float* edge_attr  = (const float*)d_in[2];
  const int*   batch      = (const int*)  d_in[3];
  const float* W_fc1      = (const float*)d_in[4];
  const float* b_fc1      = (const float*)d_in[5];
  const float* W_fc2      = (const float*)d_in[6];
  const float* b_fc2      = (const float*)d_in[7];
  const float* W_root     = (const float*)d_in[8];
  const float* b_conv     = (const float*)d_in[9];
  const float* W_pre      = (const float*)d_in[10];
  const float* b_pre      = (const float*)d_in[11];
  const float* W_out      = (const float*)d_in[12];
  const float* b_out      = (const float*)d_in[13];
  float* out = (float*)d_out;

  char* ws = (char*)d_ws;
  auto alignup = [](size_t v) { return (v + 255) & ~(size_t)255; };
  size_t off = 0;
  f16* xh = (f16*)(ws + off);
  off = alignup(off + (size_t)N_NODES * NODE_DIM * sizeof(f16));
  f16* Bfrag = (f16*)(ws + off);
  off = alignup(off + (size_t)KTILES * NT * 32 * 16 * sizeof(f16));
  f16* h16 = (f16*)(ws + off);
  off = alignup(off + (size_t)N_EDGES * HSTRIDE * sizeof(f16));
  // Contiguous f32 accumulators (zeroed each call): aggr | cnt | gsum | gcnt
  float* aggr = (float*)(ws + off);
  float* cnt  = aggr + (size_t)N_NODES * HIDDEN;
  float* gsum = cnt + N_NODES;
  float* gcnt = gsum + N_GRAPHS * HIDDEN;
  int zero_n = N_NODES * HIDDEN + N_NODES + N_GRAPHS * HIDDEN + N_GRAPHS;

  k_zero<<<2048, 256, 0, stream>>>(aggr, zero_n);
  k_x_to_f16<<<(N_NODES * NODE_DIM + 255) / 256, 256, 0, stream>>>(x, xh, N_NODES * NODE_DIM);
  k_build_bfrag<<<(KTILES * NT * 32 + 255) / 256, 256, 0, stream>>>(W_fc2, b_fc2, Bfrag);
  k_edge_fc1<<<N_EDGES, MLP_HID, 0, stream>>>(edge_attr, edge_index, W_fc1, b_fc1, h16, cnt);

  int tiles_per_block = GEMM_WAVES * GEMM_T;
  int gemm_blocks = (EDGE_TILES + tiles_per_block - 1) / tiles_per_block;
  k_edge_gemm<<<gemm_blocks, GEMM_WAVES * 32, 0, stream>>>(xh, h16, Bfrag, edge_index, aggr);

  k_node<<<N_NODES, HIDDEN, 0, stream>>>(x, batch, aggr, cnt, W_root, b_conv,
                                         W_pre, b_pre, gsum, gcnt);
  k_graph_out<<<N_GRAPHS, HIDDEN, 0, stream>>>(gsum, gcnt, W_out, b_out, out);
}
